// VanillaRNN_70025146794452
// MI455X (gfx1250) — matrix operand order repaired and checked by reading, not compiled, not run on hardware
//
#include <hip/hip_runtime.h>
#include <hip/hip_bf16.h>
#include <math.h>
#include <stdint.h>

#define SEQ_LEN    1024
#define INPUT_DIM  64
#define NUM_HIDDEN 512
#define NUM_CLASSES 10
#define BATCH      256
#define KDIM       (INPUT_DIM + NUM_HIDDEN)   // 576 = concat [x | h] reduction dim
#define KTILES     (KDIM / 32)                // 18 WMMA k-steps

typedef __attribute__((ext_vector_type(16))) __bf16 v16bf;
typedef __attribute__((ext_vector_type(8)))  __bf16 v8bf;
typedef __attribute__((ext_vector_type(8)))  float  v8f;

__device__ __forceinline__ __bf16 f2bf(float f) {
  union { float f; uint32_t u; } v; v.f = f;
  uint32_t u = v.u;
  uint32_t r16 = (u + 0x7FFFu + ((u >> 16) & 1u)) >> 16;   // round-to-nearest-even
  union { uint16_t s; __bf16 b; } o; o.s = (uint16_t)r16;
  return o.b;
}
__device__ __forceinline__ float bf2f(__bf16 b) {
  union { uint16_t s; __bf16 b; } i; i.b = b;
  union { uint32_t u; float f; } o; o.u = ((uint32_t)i.s) << 16;
  return o.f;
}
__device__ __forceinline__ v16bf cat8(v8bf lo, v8bf hi) {
  v16bf r;
#pragma unroll
  for (int i = 0; i < 8; ++i) { r[i] = lo[i]; r[i + 8] = hi[i]; }
  return r;
}

// Native CDNA5 V_TANH (single trans-VALU op, co-executes with the matrix pipe)
__device__ __forceinline__ float fast_tanh(float x) {
#if __has_builtin(__builtin_amdgcn_tanhf)
  return __builtin_amdgcn_tanhf(x);
#elif __has_builtin(__builtin_amdgcn_tanh_f32)
  return __builtin_amdgcn_tanh_f32(x);
#else
  return tanhf(x);
#endif
}

// ---- preprocessing: WT[n][k] = bf16 of [Whx; Whh] transposed, row-major over K=576 ----
__global__ void rnn_build_wt(const float* __restrict__ Whx,
                             const float* __restrict__ Whh,
                             __bf16* __restrict__ WT) {
  int i = blockIdx.x * blockDim.x + threadIdx.x;
  if (i >= NUM_HIDDEN * KDIM) return;
  int n = i / KDIM, k = i % KDIM;
  float v = (k < INPUT_DIM) ? Whx[k * NUM_HIDDEN + n]
                            : Whh[(k - INPUT_DIM) * NUM_HIDDEN + n];
  WT[i] = f2bf(v);
}

// ---- preprocessing: x f32 -> bf16, same [B][T][64] layout ----
__global__ void rnn_conv_x(const float* __restrict__ x, __bf16* __restrict__ xb, int n) {
  int i = blockIdx.x * blockDim.x + threadIdx.x;
  if (i < n) xb[i] = f2bf(x[i]);
}

// ---- main sequential scan: one WG per 16-batch-row tile, h carried in LDS ----
__global__ __launch_bounds__(512)
void rnn_scan(const __bf16* __restrict__ WT,
              const __bf16* __restrict__ xbf,
              const float*  __restrict__ bh,
              const float*  __restrict__ Wph,
              const float*  __restrict__ bp,
              float*        __restrict__ out) {
  // double-buffered activation tile: cols [0,64) = x_t, cols [64,576) = h_t
  __shared__ __bf16 Abuf[2][16][KDIM];

  const int tid    = threadIdx.x;
  const int m0     = blockIdx.x * 16;   // batch row base
  const int waveId = tid >> 5;          // 16 waves
  const int lane   = tid & 31;
  const int nLane  = lane & 15;
  const int hiHalf = lane >> 4;         // 0 or 1
  const int n0     = waveId * 32;       // this wave owns output cols [n0, n0+32)

  // init A[0]: x_0 in cols 0..63, h_0 = 0
  for (int i = tid; i < 16 * KDIM; i += 512) {
    int m = i / KDIM, k = i % KDIM;
    __bf16 v;
    if (k < INPUT_DIM) {
      v = xbf[((size_t)(m0 + m) * SEQ_LEN + 0) * INPUT_DIM + k];
    } else {
      union { uint16_t s; __bf16 b; } z; z.s = 0; v = z.b;
    }
    Abuf[0][m][k] = v;
  }
  __syncthreads();

  const float bias0 = bh[n0 + nLane];
  const float bias1 = bh[n0 + 16 + nLane];

  for (int t = 0; t < SEQ_LEN; ++t) {
    const int cur = t & 1, nxt = cur ^ 1;
    v8f acc0 = {};
    v8f acc1 = {};
    const __bf16* arow  = &Abuf[cur][nLane][0];
    const __bf16* brow0 = WT + (size_t)(n0 + nLane) * KDIM;
    const __bf16* brow1 = WT + (size_t)(n0 + 16 + nLane) * KDIM;

#pragma unroll 3
    for (int kt = 0; kt < KTILES; ++kt) {
      const int k0 = kt * 32;
      // A fragment (contiguous K: lane<16 -> K{0..15}, lane>=16 -> K{16..31})
      v8bf alo = *(const v8bf*)(arow + k0 + hiHalf * 16);
      v8bf ahi = *(const v8bf*)(arow + k0 + hiHalf * 16 + 8);
      v16bf a  = cat8(alo, ahi);
      // B fragments: lane holds column n = n0+nLane, 16 contiguous K values
      v8bf b0lo = *(const v8bf*)(brow0 + k0 + hiHalf * 16);
      v8bf b0hi = *(const v8bf*)(brow0 + k0 + hiHalf * 16 + 8);
      acc0 = __builtin_amdgcn_wmma_f32_16x16x32_bf16(
                 false, a, false, cat8(b0lo, b0hi), (short)0, acc0, false, false);
      v8bf b1lo = *(const v8bf*)(brow1 + k0 + hiHalf * 16);
      v8bf b1hi = *(const v8bf*)(brow1 + k0 + hiHalf * 16 + 8);
      acc1 = __builtin_amdgcn_wmma_f32_16x16x32_bf16(
                 false, a, false, cat8(b1lo, b1hi), (short)0, acc1, false, false);
    }

    // bias + tanh (native V_TANH), scatter h_{t+1} (bf16) into the alternate buffer
    // C/D layout: VGPR r, lane -> element (M = r + 8*(lane/16), N = lane%16)
#pragma unroll
    for (int r = 0; r < 8; ++r) {
      const int m = hiHalf * 8 + r;
      Abuf[nxt][m][INPUT_DIM + n0 + nLane]      = f2bf(fast_tanh(acc0[r] + bias0));
      Abuf[nxt][m][INPUT_DIM + n0 + 16 + nLane] = f2bf(fast_tanh(acc1[r] + bias1));
    }

    // async-prefetch x_{t+1} (16 rows x 128B) into the alternate buffer's x region
    if (t + 1 < SEQ_LEN && tid < 128) {
      const int row = tid >> 3, chunk = tid & 7;
      const __bf16* src =
          xbf + ((size_t)(m0 + row) * SEQ_LEN + (t + 1)) * INPUT_DIM + chunk * 8;
      uint32_t ldsAddr = (uint32_t)(uintptr_t)&Abuf[nxt][row][chunk * 8];
      asm volatile("global_load_async_to_lds_b128 %0, %1, off"
                   :: "v"(ldsAddr), "v"(src) : "memory");
    }
    asm volatile("s_wait_asynccnt 0x0" ::: "memory");
    __syncthreads();
  }

  // final projection p = h_final @ Wph + bp ; h_final sits in Abuf[0] (1024 is even)
  if (tid < 16 * NUM_CLASSES) {
    const int m = tid / NUM_CLASSES, c = tid % NUM_CLASSES;
    float acc = bp[c];
    for (int k = 0; k < NUM_HIDDEN; ++k)
      acc += bf2f(Abuf[0][m][INPUT_DIM + k]) * Wph[k * NUM_CLASSES + c];
    out[(size_t)(m0 + m) * NUM_CLASSES + c] = acc;
  }
}

extern "C" void kernel_launch(void* const* d_in, const int* in_sizes, int n_in,
                              void* d_out, int out_size, void* d_ws, size_t ws_size,
                              hipStream_t stream) {
  const float* x   = (const float*)d_in[0];
  const float* Whx = (const float*)d_in[1];
  const float* Whh = (const float*)d_in[2];
  const float* Wph = (const float*)d_in[3];
  const float* bh  = (const float*)d_in[4];
  const float* bp  = (const float*)d_in[5];
  float* out = (float*)d_out;

  // workspace layout: [0, 576KB) transposed bf16 weights, [1MB, 33MB) bf16 x
  __bf16* WT  = (__bf16*)d_ws;
  __bf16* xbf = (__bf16*)((char*)d_ws + (size_t)(1 << 20));

  int nwt = NUM_HIDDEN * KDIM;
  rnn_build_wt<<<(nwt + 255) / 256, 256, 0, stream>>>(Whx, Whh, WT);
  int nx = BATCH * SEQ_LEN * INPUT_DIM;
  rnn_conv_x<<<(nx + 255) / 256, 256, 0, stream>>>(x, xbf, nx);
  rnn_scan<<<BATCH / 16, 512, 0, stream>>>(WT, xbf, bh, Wph, bp, out);
}
